// Transformer_83915071030213
// MI455X (gfx1250) — matrix operand order repaired and checked by reading, not compiled, not run on hardware
//
#include <hip/hip_runtime.h>
#include <math.h>

// ---------------------------------------------------------------------------
// Types for gfx1250 WMMA (wave32)
// ---------------------------------------------------------------------------
typedef __bf16 bf16;
typedef __attribute__((ext_vector_type(4)))  __bf16 v4bf;
typedef __attribute__((ext_vector_type(8)))  __bf16 v8bf;
typedef __attribute__((ext_vector_type(16))) __bf16 v16bf;
typedef __attribute__((ext_vector_type(8)))  float  v8f;
typedef __attribute__((ext_vector_type(4)))  int    v4i;

// Model dims (match reference)
#define BB 4
#define TT 1024
#define DD 1024
#define HH 16
#define HD 64
#define LL 4
#define FFD 4096
#define VV 32000
#define MM (BB*TT)   // 4096 rows through all GEMMs

// ---------------------------------------------------------------------------
// gfx1250 async global->LDS copies (ASYNCcnt), guarded so compile never breaks.
// Builtin signature (from compiler diagnostic): (int4 AS1*, int4 AS3*, imm, imm)
// ---------------------------------------------------------------------------
#if defined(__has_builtin)
#if __has_builtin(__builtin_amdgcn_global_load_async_to_lds_b128)
#define USE_ASYNC_LDS 1
#endif
#endif

#ifdef USE_ASYNC_LDS
__device__ __forceinline__ void async_cp16(const bf16* g, bf16* l) {
    __builtin_amdgcn_global_load_async_to_lds_b128(
        (__attribute__((address_space(1))) v4i*)(v4i*)const_cast<bf16*>(g),
        (__attribute__((address_space(3))) v4i*)(v4i*)l, 0, 0);
}
__device__ __forceinline__ void wait_async0() {
#if __has_builtin(__builtin_amdgcn_s_wait_asynccnt)
    __builtin_amdgcn_s_wait_asynccnt(0);
#else
    asm volatile("s_wait_asynccnt 0x0" ::: "memory");
#endif
}
#endif

// ---------------------------------------------------------------------------
// Fragment loaders (ISA 7.12.2 layouts).
// A-operand (16x32 bf16): lane holds row m; elems 0-7 = K(half*8..+7),
// elems 8-15 = K(16+half*8..+7).  Caller passes p = rowbase + half*8.
// B-operand (32x16 bf16): lane holds col n; elems 0-7 = K(half*8..+7),
// elems 8-15 = K(16+half*8..+7).  Caller passes p = colbase + half*8.
// ---------------------------------------------------------------------------
__device__ __forceinline__ v16bf load_frag_a(const bf16* p) {
    v8bf lo = *(const v8bf*)p;
    v8bf hi = *(const v8bf*)(p + 16);
    v16bf r;
#pragma unroll
    for (int i = 0; i < 8; ++i) { r[i] = lo[i]; r[8 + i] = hi[i]; }
    return r;
}
__device__ __forceinline__ v16bf load_frag_b(const bf16* p) {
    v8bf lo = *(const v8bf*)p;
    v8bf hi = *(const v8bf*)(p + 16);
    v16bf r;
#pragma unroll
    for (int i = 0; i < 8; ++i) { r[i] = lo[i]; r[8 + i] = hi[i]; }
    return r;
}
__device__ __forceinline__ v8f vzero8() {
    v8f z;
#pragma unroll
    for (int i = 0; i < 8; ++i) z[i] = 0.f;
    return z;
}

// ---------------------------------------------------------------------------
// Embedding: x[b,t,:] = token_emb[ids[b,t],:] + pos_emb[t,:]
// ---------------------------------------------------------------------------
__global__ __launch_bounds__(256) void embed_kernel(
    const float* __restrict__ tok, const float* __restrict__ pos,
    const int* __restrict__ ids, float* __restrict__ x)
{
    const int row = blockIdx.x;             // b*T + t
    const int t   = row & (TT - 1);
    const int id  = ids[row];
    float4 a = ((const float4*)(tok + (size_t)id * DD))[threadIdx.x];
    float4 p = ((const float4*)(pos + (size_t)t  * DD))[threadIdx.x];
    a.x += p.x; a.y += p.y; a.z += p.z; a.w += p.w;
    ((float4*)(x + (size_t)row * DD))[threadIdx.x] = a;
}

// ---------------------------------------------------------------------------
// LayerNorm (f32 in, bf16 out), one 256-thread block per row of 1024.
// ---------------------------------------------------------------------------
__global__ __launch_bounds__(256) void ln_kernel(
    const float* __restrict__ x, const float* __restrict__ w,
    const float* __restrict__ b, bf16* __restrict__ out)
{
    const int row = blockIdx.x;
    const int tid = threadIdx.x;
    const float4 v = ((const float4*)(x + (size_t)row * DD))[tid];
    float s  = v.x + v.y + v.z + v.w;
    float s2 = v.x*v.x + v.y*v.y + v.z*v.z + v.w*v.w;
    __shared__ float r1[256], r2[256];
    r1[tid] = s; r2[tid] = s2;
    __syncthreads();
    for (int off = 128; off > 0; off >>= 1) {
        if (tid < off) { r1[tid] += r1[tid + off]; r2[tid] += r2[tid + off]; }
        __syncthreads();
    }
    const float mean = r1[0] * (1.0f / DD);
    const float var  = r2[0] * (1.0f / DD) - mean * mean;
    const float rstd = rsqrtf(var + 1e-5f);
    const float4 wv = ((const float4*)w)[tid];
    const float4 bv = ((const float4*)b)[tid];
    v4bf o;
    o[0] = (bf16)((v.x - mean) * rstd * wv.x + bv.x);
    o[1] = (bf16)((v.y - mean) * rstd * wv.y + bv.y);
    o[2] = (bf16)((v.z - mean) * rstd * wv.z + bv.z);
    o[3] = (bf16)((v.w - mean) * rstd * wv.w + bv.w);
    ((v4bf*)(out + (size_t)row * DD))[tid] = o;
}

// ---------------------------------------------------------------------------
// f32 -> bf16 elementwise (already-transposed weights), 4 elems/thread.
// ---------------------------------------------------------------------------
__global__ __launch_bounds__(256) void cvt_kernel(
    const float* __restrict__ in, bf16* __restrict__ out, long long n)
{
    long long i = ((long long)blockIdx.x * 256 + threadIdx.x) * 4;
    if (i >= n) return;
    float4 v = *(const float4*)(in + i);
    v4bf o;
    o[0] = (bf16)v.x; o[1] = (bf16)v.y; o[2] = (bf16)v.z; o[3] = (bf16)v.w;
    *(v4bf*)(out + i) = o;
}

// ---------------------------------------------------------------------------
// f32 [K,N] -> bf16 [N,K] tiled transpose-convert (32x32 LDS tile).
// Makes every GEMM B-stage a contiguous 16B copy (async-LDS friendly).
// ---------------------------------------------------------------------------
__global__ __launch_bounds__(256) void cvt_t_kernel(
    const float* __restrict__ in, bf16* __restrict__ out, int K, int N)
{
    __shared__ float s[32][33];
    const int k0 = blockIdx.y * 32, n0 = blockIdx.x * 32;
    const int a = threadIdx.x >> 5;   // 0..7
    const int c = threadIdx.x & 31;   // 0..31
#pragma unroll
    for (int r = 0; r < 4; ++r) {
        const int k = a + r * 8;
        s[k][c] = in[(size_t)(k0 + k) * N + n0 + c];
    }
    __syncthreads();
#pragma unroll
    for (int r = 0; r < 4; ++r) {
        const int n = a + r * 8;
        out[(size_t)(n0 + n) * K + k0 + c] = (bf16)s[c][n];
    }
}

// ---------------------------------------------------------------------------
// Split qkv f32 [B,T,3D] into q,k bf16 [B,H,T,HD] and v bf16 [B,H,HD,T].
// ---------------------------------------------------------------------------
__global__ __launch_bounds__(256) void split_qkv_kernel(
    const float* __restrict__ qkv, bf16* __restrict__ qh,
    bf16* __restrict__ kh, bf16* __restrict__ vt)
{
    const long long i = (long long)blockIdx.x * 256 + threadIdx.x;  // over B*T*D
    const int d = (int)(i & 63);
    const int h = (int)((i >> 6) & 15);
    const int t = (int)((i >> 10) & (TT - 1));
    const int b = (int)(i >> 20);
    const size_t src = (size_t)(b * TT + t) * (3 * DD) + h * HD + d;
    const int bh = b * HH + h;
    qh[((size_t)bh * TT + t) * HD + d] = (bf16)qkv[src];
    kh[((size_t)bh * TT + t) * HD + d] = (bf16)qkv[src + DD];
    vt[((size_t)bh * HD + d) * TT + t] = (bf16)qkv[src + 2 * DD];
}

// ---------------------------------------------------------------------------
// GEMM: C[M,N](f32 acc) = A_bf16[M,K] @ B_bf16 (B stored transposed: [N,K]).
// Block: 256 thr = 8 waves; tile 128(M) x 128(N), BK=32; wave tile 64x32
// (8 WMMA / wave / K-step).  Double-buffered LDS, async global->LDS staging.
// EPI: 0=f32 store, 1=+residual f32, 2=+bias,gelu -> bf16, 3=+bias+residual f32
// ---------------------------------------------------------------------------
template <int EPI>
__global__ __launch_bounds__(256) void gemm_kernel(
    const bf16* __restrict__ A, const bf16* __restrict__ Bt,
    float* __restrict__ outF, bf16* __restrict__ outB,
    const float* __restrict__ bias, const float* __restrict__ res,
    int M, int N, int K)
{
    __shared__ alignas(16) bf16 sA[2][128 * 32];    // [m][k]
    __shared__ alignas(16) bf16 sB[2][128 * 32];    // [n][k]
    const int tid  = threadIdx.x;
    const int m0   = blockIdx.y * 128;
    const int n0   = blockIdx.x * 128;
    const int wave = tid >> 5;
    const int lane = tid & 31;
    const int hl   = lane >> 4;     // half-wave
    const int lm   = lane & 15;
    const int wm   = (wave & 1) * 64;
    const int wn   = (wave >> 1) * 32;

    v8f acc[4][2];
#pragma unroll
    for (int i = 0; i < 4; ++i)
#pragma unroll
        for (int j = 0; j < 2; ++j) acc[i][j] = vzero8();

    const int nk = K >> 5;

#ifdef USE_ASYNC_LDS
    // ---- async double-buffered pipeline -----------------------------------
    auto stage = [&](int buf, int k0) {
#pragma unroll
        for (int r = 0; r < 2; ++r) {
            const int idx = tid + r * 256;          // 0..511
            const int row = idx >> 2;               // 0..127
            const int cg  = (idx & 3) * 8;          // 0,8,16,24
            async_cp16(A  + (size_t)(m0 + row) * K + k0 + cg, &sA[buf][row * 32 + cg]);
            async_cp16(Bt + (size_t)(n0 + row) * K + k0 + cg, &sB[buf][row * 32 + cg]);
        }
    };
    stage(0, 0);
    wait_async0();
    __syncthreads();
    for (int i = 0; i < nk; ++i) {
        const int cur = i & 1;
        if (i + 1 < nk) stage(1 - cur, (i + 1) * 32);
        // compute from buf cur
        v16bf af[4], bfv[2];
#pragma unroll
        for (int mt = 0; mt < 4; ++mt)
            af[mt] = load_frag_a(&sA[cur][(wm + mt * 16 + lm) * 32 + hl * 8]);
#pragma unroll
        for (int nt = 0; nt < 2; ++nt)
            bfv[nt] = load_frag_b(&sB[cur][(wn + nt * 16 + lm) * 32 + hl * 8]);
#pragma unroll
        for (int mt = 0; mt < 4; ++mt)
#pragma unroll
            for (int nt = 0; nt < 2; ++nt)
                acc[mt][nt] = __builtin_amdgcn_wmma_f32_16x16x32_bf16(
                    false, af[mt], false, bfv[nt], (short)0, acc[mt][nt], false, false);
        if (i + 1 < nk) wait_async0();
        __syncthreads();
    }
#else
    // ---- synchronous fallback (single buffer) -----------------------------
    for (int i = 0; i < nk; ++i) {
        const int k0 = i * 32;
#pragma unroll
        for (int r = 0; r < 2; ++r) {
            const int idx = tid + r * 256;
            const int row = idx >> 2;
            const int cg  = (idx & 3) * 8;
            *(uint4*)&sA[0][row * 32 + cg] =
                *(const uint4*)(A + (size_t)(m0 + row) * K + k0 + cg);
            *(uint4*)&sB[0][row * 32 + cg] =
                *(const uint4*)(Bt + (size_t)(n0 + row) * K + k0 + cg);
        }
        __syncthreads();
        v16bf af[4], bfv[2];
#pragma unroll
        for (int mt = 0; mt < 4; ++mt)
            af[mt] = load_frag_a(&sA[0][(wm + mt * 16 + lm) * 32 + hl * 8]);
#pragma unroll
        for (int nt = 0; nt < 2; ++nt)
            bfv[nt] = load_frag_b(&sB[0][(wn + nt * 16 + lm) * 32 + hl * 8]);
#pragma unroll
        for (int mt = 0; mt < 4; ++mt)
#pragma unroll
            for (int nt = 0; nt < 2; ++nt)
                acc[mt][nt] = __builtin_amdgcn_wmma_f32_16x16x32_bf16(
                    false, af[mt], false, bfv[nt], (short)0, acc[mt][nt], false, false);
        __syncthreads();
    }
#endif

    // ---- epilogue: D-layout row = v + hl*8, col = lm per 16x16 tile -------
#pragma unroll
    for (int mt = 0; mt < 4; ++mt) {
#pragma unroll
        for (int nt = 0; nt < 2; ++nt) {
#pragma unroll
            for (int v = 0; v < 8; ++v) {
                const int gr = m0 + wm + mt * 16 + hl * 8 + v;
                const int gc = n0 + wn + nt * 16 + lm;
                float val = acc[mt][nt][v];
                if constexpr (EPI == 1) {
                    val += res[(size_t)gr * N + gc];
                    outF[(size_t)gr * N + gc] = val;
                } else if constexpr (EPI == 2) {
                    val += bias[gc];
                    val = 0.5f * val * (1.0f + erff(val * 0.70710678118654752f));
                    outB[(size_t)gr * N + gc] = (bf16)val;
                } else if constexpr (EPI == 3) {
                    val += bias[gc] + res[(size_t)gr * N + gc];
                    outF[(size_t)gr * N + gc] = val;
                } else {
                    outF[(size_t)gr * N + gc] = val;
                }
            }
        }
    }
}

// ---------------------------------------------------------------------------
// Fused causal flash-attention.  One wave per (b,h, 16 query rows).
// q,k: [B,H,T,HD] bf16;  v: [B,H,HD,T] bf16;  out: [B,T,D] bf16.
// ---------------------------------------------------------------------------
__device__ __forceinline__ float hmax16(float v) {
#pragma unroll
    for (int off = 1; off < 16; off <<= 1) v = fmaxf(v, __shfl_xor(v, off, 32));
    return v;
}
__device__ __forceinline__ float hsum16(float v) {
#pragma unroll
    for (int off = 1; off < 16; off <<= 1) v += __shfl_xor(v, off, 32);
    return v;
}

__global__ __launch_bounds__(32) void attn_kernel(
    const bf16* __restrict__ qh, const bf16* __restrict__ kh,
    const bf16* __restrict__ vt, bf16* __restrict__ obf)
{
    const int bh = blockIdx.y;               // b*H + h
    const int q0 = blockIdx.x * 16;
    const int b  = bh >> 4, h = bh & 15;
    const int lane = threadIdx.x;
    const int hl = lane >> 4, lm = lane & 15;
    const float scale = 0.125f;              // 1/sqrt(64)
    const float NEG = -__builtin_inff();
    __shared__ alignas(16) bf16 sp[16 * 32]; // P bounce: D-layout -> A-layout

    const bf16* qrow = qh + ((size_t)bh * TT + q0 + lm) * HD;
    v16bf qf0 = load_frag_a(qrow + hl * 8);
    v16bf qf1 = load_frag_a(qrow + 32 + hl * 8);

    v8f oacc[4];
#pragma unroll
    for (int i = 0; i < 4; ++i) oacc[i] = vzero8();
    float mrun[8], lrun[8];
#pragma unroll
    for (int v = 0; v < 8; ++v) { mrun[v] = NEG; lrun[v] = 0.f; }

    const int qhi = q0 + 15;
    for (int j0 = 0; j0 <= qhi; j0 += 32) {
        // ---- S = Q @ K^T  (two 16-col tiles)
        v8f s0 = vzero8();
        {
            const bf16* kb = kh + ((size_t)bh * TT + j0 + lm) * HD + hl * 8;
            s0 = __builtin_amdgcn_wmma_f32_16x16x32_bf16(false, qf0, false, load_frag_b(kb),      (short)0, s0, false, false);
            s0 = __builtin_amdgcn_wmma_f32_16x16x32_bf16(false, qf1, false, load_frag_b(kb + 32), (short)0, s0, false, false);
        }
        const bool t1 = (j0 + 16) <= qhi;
        v8f s1 = vzero8();
        if (t1) {
            const bf16* kb = kh + ((size_t)bh * TT + j0 + 16 + lm) * HD + hl * 8;
            s1 = __builtin_amdgcn_wmma_f32_16x16x32_bf16(false, qf0, false, load_frag_b(kb),      (short)0, s1, false, false);
            s1 = __builtin_amdgcn_wmma_f32_16x16x32_bf16(false, qf1, false, load_frag_b(kb + 32), (short)0, s1, false, false);
        }
        // ---- scale + causal mask + online softmax
        float p0[8], p1[8], al[8];
#pragma unroll
        for (int v = 0; v < 8; ++v) {
            const int r  = q0 + hl * 8 + v;
            const int c0 = j0 + lm, c1 = j0 + 16 + lm;
            float a0 = s0[v] * scale; if (c0 > r) a0 = NEG;
            float a1 = t1 ? s1[v] * scale : NEG; if (c1 > r) a1 = NEG;
            float mx = hmax16(fmaxf(a0, a1));
            const float mn = fmaxf(mrun[v], mx);
            al[v] = expf(mrun[v] - mn);
            a0 = expf(a0 - mn); a1 = expf(a1 - mn);
            lrun[v] = lrun[v] * al[v] + hsum16(a0 + a1);
            mrun[v] = mn;
            p0[v] = a0; p1[v] = a1;
        }
#pragma unroll
        for (int nt = 0; nt < 4; ++nt)
#pragma unroll
            for (int v = 0; v < 8; ++v) oacc[nt][v] *= al[v];
        // ---- P: D-layout -> LDS -> A-layout
        __syncthreads();
#pragma unroll
        for (int v = 0; v < 8; ++v) {
            sp[(hl * 8 + v) * 32 + lm]      = (bf16)p0[v];
            sp[(hl * 8 + v) * 32 + 16 + lm] = (bf16)p1[v];
        }
        __syncthreads();
        v16bf pf = load_frag_a(&sp[lm * 32 + hl * 8]);
        // ---- O += P @ V  (V pre-transposed: contiguous kv per lane)
#pragma unroll
        for (int nt = 0; nt < 4; ++nt) {
            const bf16* vb = vt + ((size_t)bh * HD + nt * 16 + lm) * TT + j0 + hl * 8;
            oacc[nt] = __builtin_amdgcn_wmma_f32_16x16x32_bf16(
                false, pf, false, load_frag_b(vb), (short)0, oacc[nt], false, false);
        }
    }
    // ---- normalize and store [B,T,D] bf16
#pragma unroll
    for (int nt = 0; nt < 4; ++nt) {
#pragma unroll
        for (int v = 0; v < 8; ++v) {
            const int r = q0 + hl * 8 + v;
            const int c = h * HD + nt * 16 + lm;
            obf[((size_t)b * TT + r) * DD + c] = (bf16)(oacc[nt][v] / lrun[v]);
        }
    }
}

// ---------------------------------------------------------------------------
// Host driver
// ---------------------------------------------------------------------------
extern "C" void kernel_launch(void* const* d_in, const int* in_sizes, int n_in,
                              void* d_out, int out_size, void* d_ws, size_t ws_size,
                              hipStream_t stream) {
    (void)in_sizes; (void)n_in; (void)out_size; (void)ws_size;
    const float* tok  = (const float*)d_in[0];
    const float* pos  = (const float*)d_in[1];
    const float* ln1w = (const float*)d_in[2];
    const float* ln1b = (const float*)d_in[3];
    const float* qkvw = (const float*)d_in[4];
    const float* outw = (const float*)d_in[5];
    const float* ln2w = (const float*)d_in[6];
    const float* ln2b = (const float*)d_in[7];
    const float* upw  = (const float*)d_in[8];
    const float* upb  = (const float*)d_in[9];
    const float* dnw  = (const float*)d_in[10];
    const float* dnb  = (const float*)d_in[11];
    const float* lnfw = (const float*)d_in[12];
    const float* lnfb = (const float*)d_in[13];
    const int*   ids  = (const int*)d_in[14];
    float* logits = (float*)d_out;

    // workspace carve-up (256B aligned)
    char* p = (char*)d_ws;
    auto carve = [&](size_t bytes) { void* r = (void*)p; p += (bytes + 255) & ~(size_t)255; return r; };
    float* x    = (float*)carve((size_t)MM * DD * 4);        // residual stream
    bf16*  hbf  = (bf16*) carve((size_t)MM * DD * 2);        // LN output
    float* qkv  = (float*)carve((size_t)MM * 3 * DD * 4);
    bf16*  qh   = (bf16*) carve((size_t)MM * DD * 2);
    bf16*  kh   = (bf16*) carve((size_t)MM * DD * 2);
    bf16*  vt   = (bf16*) carve((size_t)MM * DD * 2);
    bf16*  obf  = (bf16*) carve((size_t)MM * DD * 2);
    bf16*  ffbf = (bf16*) carve((size_t)MM * FFD * 2);
    bf16*  wbuf = (bf16*) carve((size_t)VV * DD * 2);        // largest weight (token_emb)

    const dim3 blk256(256), blk32(32);

    // x = token_emb[ids] + pos_emb
    embed_kernel<<<dim3(MM), blk256, 0, stream>>>(tok, pos, ids, x);

    for (int l = 0; l < LL; ++l) {
        // h = LN1(x) -> bf16
        ln_kernel<<<dim3(MM), blk256, 0, stream>>>(x, ln1w + l * DD, ln1b + l * DD, hbf);
        // qkv = h @ qkv_w   (weight -> bf16 [N,K])
        cvt_t_kernel<<<dim3(3 * DD / 32, DD / 32), blk256, 0, stream>>>(
            qkvw + (size_t)l * DD * 3 * DD, wbuf, DD, 3 * DD);
        gemm_kernel<0><<<dim3(3 * DD / 128, MM / 128), blk256, 0, stream>>>(
            hbf, wbuf, qkv, nullptr, nullptr, nullptr, MM, 3 * DD, DD);
        // split heads (v transposed)
        split_qkv_kernel<<<dim3((MM * DD) / 256), blk256, 0, stream>>>(qkv, qh, kh, vt);
        // fused causal attention
        attn_kernel<<<dim3(TT / 16, BB * HH), blk32, 0, stream>>>(qh, kh, vt, obf);
        // x = x + attn_out @ out_w
        cvt_t_kernel<<<dim3(DD / 32, DD / 32), blk256, 0, stream>>>(
            outw + (size_t)l * DD * DD, wbuf, DD, DD);
        gemm_kernel<1><<<dim3(DD / 128, MM / 128), blk256, 0, stream>>>(
            obf, wbuf, x, nullptr, nullptr, x, MM, DD, DD);
        // h2 = LN2(x);  ff = gelu(h2 @ up_w + up_b) -> bf16
        ln_kernel<<<dim3(MM), blk256, 0, stream>>>(x, ln2w + l * DD, ln2b + l * DD, hbf);
        cvt_t_kernel<<<dim3(FFD / 32, DD / 32), blk256, 0, stream>>>(
            upw + (size_t)l * DD * FFD, wbuf, DD, FFD);
        gemm_kernel<2><<<dim3(FFD / 128, MM / 128), blk256, 0, stream>>>(
            hbf, wbuf, nullptr, ffbf, upb + l * FFD, nullptr, MM, FFD, DD);
        // x = x + ff @ down_w + down_b
        cvt_t_kernel<<<dim3(DD / 32, FFD / 32), blk256, 0, stream>>>(
            dnw + (size_t)l * FFD * DD, wbuf, FFD, DD);
        gemm_kernel<3><<<dim3(DD / 128, MM / 128), blk256, 0, stream>>>(
            ffbf, wbuf, x, nullptr, dnb + l * DD, x, MM, DD, FFD);
    }

    // final LN + tied lm_head: logits = LN(x) @ token_emb^T
    // token_emb is [V,D] = [N,K] already -> plain convert, no transpose.
    ln_kernel<<<dim3(MM), blk256, 0, stream>>>(x, lnfw, lnfb, hbf);
    cvt_kernel<<<dim3((VV * DD) / 1024), blk256, 0, stream>>>(
        tok, wbuf, (long long)VV * DD);
    gemm_kernel<0><<<dim3(VV / 128, MM / 128), blk256, 0, stream>>>(
        hbf, wbuf, logits, nullptr, nullptr, nullptr, MM, VV, DD);
}
